// SurfNetwork_3049426780606
// MI455X (gfx1250) — compile-verified, matrix operand order checked by reading, not run on hardware
//
#include <hip/hip_runtime.h>
#include <hip/hip_bf16.h>

typedef __attribute__((ext_vector_type(16))) _Float16 v16h;
typedef __attribute__((ext_vector_type(8)))  _Float16 v8h;
typedef __attribute__((ext_vector_type(8)))  float    v8f;

#define NB    16384   // B
#define NS    128     // N samples per batch
#define NL    6       // levels
#define ND    4       // feat dim per table row
#define NDIR  16      // direction dims
#define NH    64      // hidden
#define KPAD  64      // padded input K (34 -> 64)
#define STR   72      // LDS row stride in halfs (144B: multiple of 16B, conflict-skewed)

// Build a 16x16x32 f16 A/B fragment from an LDS row (layout [row][K], stride STR).
// lane&15 selects the matrix row (A) / column (B); lane>>4 selects which K-half.
__device__ __forceinline__ v16h load_frag(const _Float16* rowp, int K0, int hi) {
    v8h lo = *(const v8h*)(rowp + K0 + hi * 8);
    v8h hh = *(const v8h*)(rowp + K0 + 16 + hi * 8);
    return __builtin_shufflevector(lo, hh, 0,1,2,3,4,5,6,7,8,9,10,11,12,13,14,15);
}

__device__ __forceinline__ v8f wmma16(v16h a, v16h b, v8f c) {
    // (neg_a, A, neg_b, B, c_mod, C, reuse_a, reuse_b)
    return __builtin_amdgcn_wmma_f32_16x16x32_f16(false, a, false, b, (short)0, c, false, false);
}

__global__ __launch_bounds__(256)
void surfnet_kernel(const int* __restrict__ x, const float* __restrict__ dir,
                    const float* __restrict__ gridW,
                    const float* __restrict__ W0, const float* __restrict__ b0,
                    const float* __restrict__ W1, const float* __restrict__ b1,
                    const float* __restrict__ W2, const float* __restrict__ b2,
                    float* __restrict__ outSigma, float* __restrict__ outColor) {
    __shared__ __align__(16) _Float16 sAct[NS * STR];       // activations, f16, [sample][K]
    __shared__ __align__(16) _Float16 sW0[NH * STR];        // W0^T  [out][in], in padded 34->64
    __shared__ __align__(16) _Float16 sW1[NH * STR];        // W1^T  [out][in]
    __shared__ __align__(16) _Float16 sW2[16 * STR];        // W2^T  [out(pad 3->16)][in]
    __shared__ __align__(16) _Float16 sH[8 * 16 * STR];     // per-wave hidden restage
    __shared__ float sSig0[NS];
    __shared__ float sSig[NS];
    __shared__ float sCol[NS * 3];
    __shared__ float sRed[NS * 3];
    __shared__ float sMax[64];

    const int b   = blockIdx.x;
    const int tid = threadIdx.x;

    // ---- Phase 1a: gather + density + input assembly (threads 0..127: one sample each)
    if (tid < NS) {
        const int n = tid;
        const int* xp = x + ((long)b * NS + n) * NL;
        _Float16* arow = sAct + n * STR;
        const float* dp = dir + (long)b * NDIR;
        #pragma unroll
        for (int j = 0; j < NDIR; ++j) arow[j] = (_Float16)dp[j];
        float prod = 1.0f;
        #pragma unroll
        for (int l = 0; l < NL; ++l) {
            const int id = xp[l];
            const float4 row = *(const float4*)(gridW + (long)id * ND);   // global_load_b128, L2-resident
            prod *= fminf(fmaxf(row.x, 0.0f), 1.0f);
            arow[NDIR + l * 3 + 0] = (_Float16)row.y;
            arow[NDIR + l * 3 + 1] = (_Float16)row.z;
            arow[NDIR + l * 3 + 2] = (_Float16)row.w;
        }
        #pragma unroll
        for (int j = NDIR + NL * 3; j < KPAD; ++j) arow[j] = (_Float16)0.0f;
        sSig0[n] = prod + 1e-4f;
    }
    // ---- Phase 1b: stage weights transposed into LDS as f16 (all 256 threads)
    for (int i = tid; i < NH * NH; i += 256) {               // W0: (34,64) -> padded 64x64
        int k = i >> 6, j = i & 63;
        sW0[j * STR + k] = (_Float16)((k < (NDIR + NL * 3)) ? W0[k * NH + j] : 0.0f);
    }
    for (int i = tid; i < NH * NH; i += 256) {               // W1: (64,64)
        int k = i >> 6, j = i & 63;
        sW1[j * STR + k] = (_Float16)W1[k * NH + j];
    }
    for (int i = tid; i < NH * 16; i += 256) {               // W2: (64,3) -> padded 64x16
        int k = i >> 4, j = i & 15;
        sW2[j * STR + k] = (_Float16)((j < 3) ? W2[k * 3 + j] : 0.0f);
    }
    __syncthreads();

    // ---- Phase 2: MLP via WMMA; one 16-sample tile per wave (8 waves)
    {
        const int lane = tid & 31;
        const int row  = lane & 15;   // A-row / B-col / C-col
        const int hi   = lane >> 4;   // K-half select; C-row-half select
        const int base = (tid >> 5) * 16;
        _Float16* hbuf = sH + (tid >> 5) * 16 * STR;

        // Layer 0: (pad64) -> 64, ReLU
        v16h a0 = load_frag(sAct + (base + row) * STR, 0,  hi);
        v16h a1 = load_frag(sAct + (base + row) * STR, 32, hi);
        #pragma unroll
        for (int t = 0; t < 4; ++t) {
            const float bv = b0[t * 16 + row];
            v8f c;
            #pragma unroll
            for (int r = 0; r < 8; ++r) c[r] = bv;           // bias pre-load (col depends only on lane)
            c = wmma16(a0, load_frag(sW0 + (t * 16 + row) * STR, 0,  hi), c);
            c = wmma16(a1, load_frag(sW0 + (t * 16 + row) * STR, 32, hi), c);
            #pragma unroll
            for (int r = 0; r < 8; ++r)                       // C-layout -> [sample][feature] restage
                hbuf[(hi * 8 + r) * STR + t * 16 + row] = (_Float16)fmaxf(c[r], 0.0f);
        }

        // Layer 1: 64 -> 64, ReLU (reads fully before overwriting hbuf)
        a0 = load_frag(hbuf + row * STR, 0,  hi);
        a1 = load_frag(hbuf + row * STR, 32, hi);
        #pragma unroll
        for (int t = 0; t < 4; ++t) {
            const float bv = b1[t * 16 + row];
            v8f c;
            #pragma unroll
            for (int r = 0; r < 8; ++r) c[r] = bv;
            c = wmma16(a0, load_frag(sW1 + (t * 16 + row) * STR, 0,  hi), c);
            c = wmma16(a1, load_frag(sW1 + (t * 16 + row) * STR, 32, hi), c);
            #pragma unroll
            for (int r = 0; r < 8; ++r)
                hbuf[(hi * 8 + r) * STR + t * 16 + row] = (_Float16)fmaxf(c[r], 0.0f);
        }

        // Layer 2: 64 -> 3 (padded 16), sigmoid
        a0 = load_frag(hbuf + row * STR, 0,  hi);
        a1 = load_frag(hbuf + row * STR, 32, hi);
        {
            const float bv = (row < 3) ? b2[row] : 0.0f;
            v8f c;
            #pragma unroll
            for (int r = 0; r < 8; ++r) c[r] = bv;
            c = wmma16(a0, load_frag(sW2 + row * STR, 0,  hi), c);
            c = wmma16(a1, load_frag(sW2 + row * STR, 32, hi), c);
            if (row < 3) {
                #pragma unroll
                for (int r = 0; r < 8; ++r)
                    sCol[(base + hi * 8 + r) * 3 + row] = 1.0f / (1.0f + __expf(-c[r]));
            }
        }
    }
    __syncthreads();

    // ---- Phase 3: normalize sigma by per-b max, composite, reduce colors over N
    if (tid < 64) sMax[tid] = fmaxf(sSig0[tid], sSig0[tid + 64]);
    __syncthreads();
    for (int off = 32; off >= 1; off >>= 1) {
        if (tid < off) sMax[tid] = fmaxf(sMax[tid], sMax[tid + off]);
        __syncthreads();
    }
    const float inv = 1.0f / sMax[0];
    if (tid < NS) {
        const float s = sSig0[tid] * inv;
        sSig[tid] = s;
        outSigma[(long)b * NS + tid] = s;
    }
    __syncthreads();
    if (tid < NS) {
        const float w = ((tid == 0) ? 1.0f : (1.0f - sSig[tid - 1])) * sSig[tid];
        sRed[tid * 3 + 0] = w * sCol[tid * 3 + 0];
        sRed[tid * 3 + 1] = w * sCol[tid * 3 + 1];
        sRed[tid * 3 + 2] = w * sCol[tid * 3 + 2];
    }
    __syncthreads();
    for (int off = 64; off >= 1; off >>= 1) {
        if (tid < off) {
            sRed[tid * 3 + 0] += sRed[(tid + off) * 3 + 0];
            sRed[tid * 3 + 1] += sRed[(tid + off) * 3 + 1];
            sRed[tid * 3 + 2] += sRed[(tid + off) * 3 + 2];
        }
        __syncthreads();
    }
    if (tid == 0) {
        outColor[b * 3 + 0] = sRed[0];
        outColor[b * 3 + 1] = sRed[1];
        outColor[b * 3 + 2] = sRed[2];
    }
}

extern "C" void kernel_launch(void* const* d_in, const int* in_sizes, int n_in,
                              void* d_out, int out_size, void* d_ws, size_t ws_size,
                              hipStream_t stream) {
    const int*   x   = (const int*)d_in[0];
    const float* dd  = (const float*)d_in[1];
    const float* gw  = (const float*)d_in[2];
    const float* W0  = (const float*)d_in[3];
    const float* b0  = (const float*)d_in[4];
    const float* W1  = (const float*)d_in[5];
    const float* b1  = (const float*)d_in[6];
    const float* W2  = (const float*)d_in[7];
    const float* b2  = (const float*)d_in[8];
    float* outSigma = (float*)d_out;                       // (B, N, 1) flat
    float* outColor = (float*)d_out + (long)NB * NS;       // (B, 3) flat
    surfnet_kernel<<<NB, 256, 0, stream>>>(x, dd, gw, W0, b0, W1, b1, W2, b2,
                                           outSigma, outColor);
}
